// PsvCrossAttention_14104672600566
// MI455X (gfx1250) — matrix operand order repaired and checked by reading, not compiled, not run on hardware
//
#include <hip/hip_runtime.h>
#include <stdint.h>

typedef __attribute__((ext_vector_type(16))) __bf16 v16bf;
typedef __attribute__((ext_vector_type(8)))  float  v8f;

union Frag16 { uint4 u4[2]; v16bf v; };
union Pack8  { unsigned short s[8]; uint4 u; };

__device__ __forceinline__ unsigned short f32_bf16(float f) {
  unsigned u = __float_as_uint(f);
  u += 0x7fffu + ((u >> 16) & 1u);
  return (unsigned short)(u >> 16);
}

// ---------------------------------------------------------------- converts
__global__ void cvt_f32_bf16(const float* __restrict__ in,
                             unsigned short* __restrict__ out, int n) {
  int i = blockIdx.x * blockDim.x + threadIdx.x;
  if (i < n) out[i] = f32_bf16(in[i]);
}

// [B][C][N] f32 -> [B][N][C] bf16 (tiled 32x32 transpose through LDS)
__global__ void __launch_bounds__(256)
cvt_transpose_bf16(const float* __restrict__ in, unsigned short* __restrict__ out,
                   int C, int N) {
  __shared__ float tile[32][33];
  const int b = blockIdx.z;
  const int n0 = blockIdx.x * 32, c0 = blockIdx.y * 32;
#pragma unroll
  for (int i = 0; i < 4; ++i) {
    int c = c0 + threadIdx.y + i * 8;
    tile[threadIdx.y + i * 8][threadIdx.x] =
        in[((long long)b * C + c) * N + n0 + threadIdx.x];
  }
  __syncthreads();
#pragma unroll
  for (int i = 0; i < 4; ++i) {
    int n = n0 + threadIdx.y + i * 8;
    out[((long long)b * N + n) * C + c0 + threadIdx.x] =
        f32_bf16(tile[threadIdx.x][threadIdx.y + i * 8]);
  }
}

// ---------------------------------------------------------------- GEMM
// C[M,N] = A[M,K] * BT[N,K]^T (bf16 in, f32 accum). 256 threads = 8 waves,
// block tile 128x128, wave tile 32x64 (2x4 WMMA frags), K-step 32,
// register-pipelined staging (all b128, no LDS transposes).
// mode 0: bf16 out[m*N+n]                              ([h][d][n])
// mode 1: bf16 out[((m>>6)*N+n)*64+(m&63)] * scale     ([h][n][d], b128 stores)
// mode 2: f32  out[m*N+n] + bias[m]
__global__ void __launch_bounds__(256)
gemm_bf16_wmma(const unsigned short* __restrict__ A,
               const unsigned short* __restrict__ BT,
               void* __restrict__ Out,
               const float* __restrict__ bias,
               int M, int K, int N,
               long long strideB, long long strideO,
               float scale, int mode)
{
  __shared__ alignas(16) unsigned short Asl[128][40];  // [m][k]
  __shared__ alignas(16) unsigned short Bsl[128][40];  // [n][k]

  const int tid = threadIdx.x;
  const int lane = tid & 31;
  const int w = tid >> 5, wm = w >> 1, wn = w & 1;
  const int l15 = lane & 15, lh = lane >> 4;
  const int n0 = blockIdx.x * 128, m0 = blockIdx.y * 128, b = blockIdx.z;
  const unsigned short* Bb = BT + (long long)b * strideB;

  // staging coords: 512 uint4 per tile, thread t covers linear {2t, 2t+1}
  const int r0s = (tid * 2) >> 2, c0s = (tid * 2) & 3;        // it=0
  const int r1s = (tid * 2 + 1) >> 2, c1s = (tid * 2 + 1) & 3; // it=1

  auto ldA = [&](int kt, int row, int c8) -> uint4 {
    if (m0 + row < M)
      return *(const uint4*)(A + (long long)(m0 + row) * K + kt + c8 * 8);
    return make_uint4(0u, 0u, 0u, 0u);
  };
  auto ldB = [&](int kt, int row, int c8) -> uint4 {
    return *(const uint4*)(Bb + (long long)(n0 + row) * K + kt + c8 * 8);
  };

  v8f zero = {};
  v8f acc[2][4];
#pragma unroll
  for (int i = 0; i < 2; ++i)
#pragma unroll
    for (int j = 0; j < 4; ++j) acc[i][j] = zero;

  uint4 ra0 = ldA(0, r0s, c0s), ra1 = ldA(0, r1s, c1s);
  uint4 rb0 = ldB(0, r0s, c0s), rb1 = ldB(0, r1s, c1s);

  for (int kt = 0; kt < K; kt += 32) {
    __syncthreads();
    *(uint4*)&Asl[r0s][c0s * 8] = ra0;
    *(uint4*)&Asl[r1s][c1s * 8] = ra1;
    *(uint4*)&Bsl[r0s][c0s * 8] = rb0;
    *(uint4*)&Bsl[r1s][c1s * 8] = rb1;
    __syncthreads();
    if (kt + 32 < K) {  // issue next tile's global loads before compute
      ra0 = ldA(kt + 32, r0s, c0s); ra1 = ldA(kt + 32, r1s, c1s);
      rb0 = ldB(kt + 32, r0s, c0s); rb1 = ldB(kt + 32, r1s, c1s);
    }

    Frag16 a[2], bf[4];
#pragma unroll
    for (int mi = 0; mi < 2; ++mi) {
      int m = wm * 32 + mi * 16 + l15;
      int k0 = lh * 8;                 // A: elems 0..7 = K k0.., 8..15 = K k0+16..
      a[mi].u4[0] = *(const uint4*)&Asl[m][k0];
      a[mi].u4[1] = *(const uint4*)&Asl[m][k0 + 16];
    }
#pragma unroll
    for (int ni = 0; ni < 4; ++ni) {
      int n = wn * 64 + ni * 16 + l15;
      int k0 = lh * 8;                 // B: elems 0..7 = K k0.., 8..15 = K k0+16..
      bf[ni].u4[0] = *(const uint4*)&Bsl[n][k0];
      bf[ni].u4[1] = *(const uint4*)&Bsl[n][k0 + 16];
    }
#pragma unroll
    for (int mi = 0; mi < 2; ++mi)
#pragma unroll
      for (int ni = 0; ni < 4; ++ni)
        acc[mi][ni] = __builtin_amdgcn_wmma_f32_16x16x32_bf16(
            false, a[mi].v, false, bf[ni].v, (short)0, acc[mi][ni], false, false);
  }

  // epilogue (C layout: lane = col, reg r = row lh*8+r)
  long long ob = (long long)b * strideO;
#pragma unroll
  for (int mi = 0; mi < 2; ++mi) {
#pragma unroll
    for (int ni = 0; ni < 4; ++ni) {
      int n_g = n0 + wn * 64 + ni * 16 + l15;
      int mbase = m0 + wm * 32 + mi * 16 + lh * 8;
      if (mode == 1) {  // 8 consecutive d within one head -> one b128 store
        Pack8 pk;
#pragma unroll
        for (int r = 0; r < 8; ++r) pk.s[r] = f32_bf16(acc[mi][ni][r] * scale);
        long long idx = ((long long)(mbase >> 6) * N + n_g) * 64 + (mbase & 63);
        *(uint4*)((unsigned short*)Out + ob + idx) = pk.u;
      } else if (mode == 2) {
#pragma unroll
        for (int r = 0; r < 8; ++r) {
          int m_g = mbase + r;
          if (m_g < M)
            ((float*)Out)[ob + (long long)m_g * N + n_g] =
                acc[mi][ni][r] * scale + bias[m_g];
        }
      } else {
#pragma unroll
        for (int r = 0; r < 8; ++r) {
          int m_g = mbase + r;
          if (m_g < M)
            ((unsigned short*)Out)[ob + (long long)m_g * N + n_g] =
                f32_bf16(acc[mi][ni][r] * scale);
        }
      }
    }
  }
}

// ---------------------------------------------------------------- flash attention
// Q: [BH][N][64] (pre-scaled), Kh: [BH][N][64], V: [BH][64][N],
// O: [B][N][512] (head h occupies cols h*64..h*64+63).
// 256 threads = 8 waves, 16 query rows per wave, streams 64-key chunks with
// online softmax. No transposes: all staging b128, all frags ds_load_b128.
__global__ void __launch_bounds__(256)
flash_attn_bf16(const unsigned short* __restrict__ Q,
                const unsigned short* __restrict__ Kh,
                const unsigned short* __restrict__ V,
                unsigned short* __restrict__ O, int N)
{
  __shared__ alignas(16) unsigned short Kc[64][72];     // [j][d]
  __shared__ alignas(16) unsigned short Vc[64][72];     // [d][j]
  __shared__ alignas(16) unsigned short Pc[8][16][72];  // per-wave [i][j] / [i][d]

  const int tid = threadIdx.x;
  const int lane = tid & 31;
  const int w = tid >> 5;
  const int l15 = lane & 15, lh = lane >> 4;

  const int bh = blockIdx.y;
  const int i0 = blockIdx.x * 128;

  const unsigned short* Qb = Q + (long long)bh * N * 64;
  const unsigned short* Kb = Kh + (long long)bh * N * 64;
  const unsigned short* Vb = V + (long long)bh * 64 * N;
  unsigned short* Ob = O + (long long)(bh >> 3) * N * 512 + (bh & 7) * 64;

  // staging coords: 512 uint4 per 64x64 tile, thread covers linear {2t,2t+1}
  const int sr0 = (tid * 2) >> 3, sc0 = (tid * 2) & 7;
  const int sr1 = (tid * 2 + 1) >> 3, sc1 = (tid * 2 + 1) & 7;
  auto ldK = [&](int j0, int row, int c8) -> uint4 {  // K rows j, d-contiguous
    return *(const uint4*)(Kb + (long long)(j0 + row) * 64 + c8 * 8);
  };
  auto ldV = [&](int j0, int row, int c8) -> uint4 {  // V rows d, j-contiguous
    return *(const uint4*)(Vb + (long long)row * N + j0 + c8 * 8);
  };

  // Q fragments stay in registers for the whole kernel (A layout).
  Frag16 aq[2];
  {
    int i = i0 + w * 16 + l15;
    int k0 = lh * 8;
#pragma unroll
    for (int kk = 0; kk < 2; ++kk) {
      aq[kk].u4[0] = *(const uint4*)(Qb + (long long)i * 64 + kk * 32 + k0);
      aq[kk].u4[1] = *(const uint4*)(Qb + (long long)i * 64 + kk * 32 + k0 + 16);
    }
  }

  v8f zero = {};
  v8f oacc[4];
#pragma unroll
  for (int dt = 0; dt < 4; ++dt) oacc[dt] = zero;
  float mrow[8], lrow[8];
#pragma unroll
  for (int r = 0; r < 8; ++r) { mrow[r] = -1e30f; lrow[r] = 0.f; }

  uint4 rk0 = ldK(0, sr0, sc0), rk1 = ldK(0, sr1, sc1);
  uint4 rv0 = ldV(0, sr0, sc0), rv1 = ldV(0, sr1, sc1);

  for (int j0 = 0; j0 < N; j0 += 64) {
    __syncthreads();
    *(uint4*)&Kc[sr0][sc0 * 8] = rk0;
    *(uint4*)&Kc[sr1][sc1 * 8] = rk1;
    *(uint4*)&Vc[sr0][sc0 * 8] = rv0;
    *(uint4*)&Vc[sr1][sc1 * 8] = rv1;
    __syncthreads();
    if (j0 + 64 < N) {  // issue next chunk's global loads before compute
      rk0 = ldK(j0 + 64, sr0, sc0); rk1 = ldK(j0 + 64, sr1, sc1);
      rv0 = ldV(j0 + 64, sr0, sc0); rv1 = ldV(j0 + 64, sr1, sc1);
    }
    if (j0 + 128 < N) {  // warm L2 for chunk after next
      __builtin_prefetch(Kb + (long long)(j0 + 128 + sr0) * 64, 0, 1);
      __builtin_prefetch(Vb + (long long)sr0 * N + j0 + 128, 0, 1);
    }

    // S = Q * K  (16 x 64 per wave)
    v8f s[4];
#pragma unroll
    for (int jt = 0; jt < 4; ++jt) s[jt] = zero;
#pragma unroll
    for (int kk = 0; kk < 2; ++kk) {
#pragma unroll
      for (int jt = 0; jt < 4; ++jt) {
        Frag16 bk;
        int j = jt * 16 + l15;
        int k0 = kk * 32 + lh * 8;
        bk.u4[0] = *(const uint4*)&Kc[j][k0];
        bk.u4[1] = *(const uint4*)&Kc[j][k0 + 16];
        s[jt] = __builtin_amdgcn_wmma_f32_16x16x32_bf16(
            false, aq[kk].v, false, bk.v, (short)0, s[jt], false, false);
      }
    }

    // online softmax: lane holds cols jt*16+l15, rows lh*8+r
    float mnew[8], psum[8];
#pragma unroll
    for (int r = 0; r < 8; ++r) {
      float mx = fmaxf(fmaxf(s[0][r], s[1][r]), fmaxf(s[2][r], s[3][r]));
#pragma unroll
      for (int sh = 1; sh < 16; sh <<= 1) mx = fmaxf(mx, __shfl_xor(mx, sh, 16));
      mnew[r] = fmaxf(mrow[r], mx);
    }
#pragma unroll
    for (int r = 0; r < 8; ++r) {
      float sum = 0.f;
#pragma unroll
      for (int jt = 0; jt < 4; ++jt) {
        float p = __expf(s[jt][r] - mnew[r]);
        s[jt][r] = p;
        sum += p;
      }
#pragma unroll
      for (int sh = 1; sh < 16; sh <<= 1) sum += __shfl_xor(sum, sh, 16);
      psum[r] = sum;
    }
#pragma unroll
    for (int r = 0; r < 8; ++r) {
      float corr = __expf(mrow[r] - mnew[r]);
      lrow[r] = lrow[r] * corr + psum[r];
      mrow[r] = mnew[r];
#pragma unroll
      for (int dt = 0; dt < 4; ++dt) oacc[dt][r] *= corr;
    }

    // C-layout -> A-layout via per-wave LDS region
#pragma unroll
    for (int jt = 0; jt < 4; ++jt)
#pragma unroll
      for (int r = 0; r < 8; ++r)
        Pc[w][lh * 8 + r][jt * 16 + l15] = f32_bf16(s[jt][r]);

    // O += P * V
#pragma unroll
    for (int kk = 0; kk < 2; ++kk) {
      Frag16 ap;
      int k0a = kk * 32 + lh * 8;
      ap.u4[0] = *(const uint4*)&Pc[w][l15][k0a];
      ap.u4[1] = *(const uint4*)&Pc[w][l15][k0a + 16];
#pragma unroll
      for (int dt = 0; dt < 4; ++dt) {
        Frag16 bv;
        int d = dt * 16 + l15;
        int k0 = kk * 32 + lh * 8;
        bv.u4[0] = *(const uint4*)&Vc[d][k0];
        bv.u4[1] = *(const uint4*)&Vc[d][k0 + 16];
        oacc[dt] = __builtin_amdgcn_wmma_f32_16x16x32_bf16(
            false, ap.v, false, bv.v, (short)0, oacc[dt], false, false);
      }
    }
  }

  // normalize, transpose wave tile through LDS, write [n][hidden] with b128
#pragma unroll
  for (int dt = 0; dt < 4; ++dt)
#pragma unroll
    for (int r = 0; r < 8; ++r)
      Pc[w][lh * 8 + r][dt * 16 + l15] = f32_bf16(oacc[dt][r] / lrow[r]);
#pragma unroll
  for (int t = 0; t < 4; ++t) {
    int linear = lane * 4 + t;          // 0..127
    int row = linear >> 3, c8 = linear & 7;
    uint4 u = *(const uint4*)&Pc[w][row][c8 * 8];
    *(uint4*)(Ob + (long long)(i0 + w * 16 + row) * 512 + c8 * 8) = u;
  }
}

// ---------------------------------------------------------------- launch
extern "C" void kernel_launch(void* const* d_in, const int* in_sizes, int n_in,
                              void* d_out, int out_size, void* d_ws, size_t ws_size,
                              hipStream_t stream) {
  (void)in_sizes; (void)n_in; (void)out_size; (void)ws_size;
  const float* seq1 = (const float*)d_in[0];
  const float* seq2 = (const float*)d_in[1];
  const float* Wq1  = (const float*)d_in[2];
  const float* Wk1  = (const float*)d_in[3];
  const float* Wv1  = (const float*)d_in[4];
  const float* Wo1  = (const float*)d_in[5];
  const float* bo1  = (const float*)d_in[6];
  const float* Wq2  = (const float*)d_in[7];
  const float* Wk2  = (const float*)d_in[8];
  const float* Wv2  = (const float*)d_in[9];
  const float* Wo2  = (const float*)d_in[10];
  const float* bo2  = (const float*)d_in[11];

  const int Bn = 2, C1 = 256, C2 = 320, H = 512, N = 2304;
  const float SCALE = 0.125f;  // 64^-0.5

  char* ws = (char*)d_ws;
  size_t off = 0;
  auto alloc = [&](size_t bytes) -> void* {
    void* p = ws + off;
    off += (bytes + 255) & ~(size_t)255;
    return p;
  };
  unsigned short* x1t  = (unsigned short*)alloc((size_t)Bn * N * C1 * 2); // [b][n][c]
  unsigned short* x2t  = (unsigned short*)alloc((size_t)Bn * N * C2 * 2);
  unsigned short* wq1b = (unsigned short*)alloc((size_t)H * C1 * 2);
  unsigned short* wk1b = (unsigned short*)alloc((size_t)H * C1 * 2);
  unsigned short* wv1b = (unsigned short*)alloc((size_t)H * C1 * 2);
  unsigned short* wo1b = (unsigned short*)alloc((size_t)C1 * H * 2);
  unsigned short* wq2b = (unsigned short*)alloc((size_t)H * C2 * 2);
  unsigned short* wk2b = (unsigned short*)alloc((size_t)H * C2 * 2);
  unsigned short* wv2b = (unsigned short*)alloc((size_t)H * C2 * 2);
  unsigned short* wo2b = (unsigned short*)alloc((size_t)C2 * H * 2);
  unsigned short* q1 = (unsigned short*)alloc((size_t)Bn * H * N * 2);  // [b][h][n][d]
  unsigned short* k1 = (unsigned short*)alloc((size_t)Bn * H * N * 2);  // [b][h][n][d]
  unsigned short* v1 = (unsigned short*)alloc((size_t)Bn * H * N * 2);  // [b][h][d][n]
  unsigned short* q2 = (unsigned short*)alloc((size_t)Bn * H * N * 2);
  unsigned short* k2 = (unsigned short*)alloc((size_t)Bn * H * N * 2);
  unsigned short* v2 = (unsigned short*)alloc((size_t)Bn * H * N * 2);
  unsigned short* o1 = (unsigned short*)alloc((size_t)Bn * N * H * 2);  // [b][n][512]
  unsigned short* o2 = (unsigned short*)alloc((size_t)Bn * N * H * 2);

  auto cv = [&](const float* src, unsigned short* dst, int n) {
    cvt_f32_bf16<<<(n + 255) / 256, 256, 0, stream>>>(src, dst, n);
  };
  cvt_transpose_bf16<<<dim3(N / 32, C1 / 32, Bn), dim3(32, 8), 0, stream>>>(seq1, x1t, C1, N);
  cvt_transpose_bf16<<<dim3(N / 32, C2 / 32, Bn), dim3(32, 8), 0, stream>>>(seq2, x2t, C2, N);
  cv(Wq1, wq1b, H * C1); cv(Wk1, wk1b, H * C1); cv(Wv1, wv1b, H * C1);
  cv(Wo1, wo1b, C1 * H);
  cv(Wq2, wq2b, H * C2); cv(Wk2, wk2b, H * C2); cv(Wv2, wv2b, H * C2);
  cv(Wo2, wo2b, C2 * H);

  dim3 blk(256);
  dim3 gp(N / 128, H / 128, Bn);  // projections: 512 x {256,320} x 2304
  gemm_bf16_wmma<<<gp, blk, 0, stream>>>(wq1b, x1t, q1, nullptr, H, C1, N,
      (long long)N * C1, (long long)H * N, SCALE, 1);
  gemm_bf16_wmma<<<gp, blk, 0, stream>>>(wk1b, x1t, k1, nullptr, H, C1, N,
      (long long)N * C1, (long long)H * N, 1.f, 1);
  gemm_bf16_wmma<<<gp, blk, 0, stream>>>(wv1b, x1t, v1, nullptr, H, C1, N,
      (long long)N * C1, (long long)H * N, 1.f, 0);
  gemm_bf16_wmma<<<gp, blk, 0, stream>>>(wq2b, x2t, q2, nullptr, H, C2, N,
      (long long)N * C2, (long long)H * N, SCALE, 1);
  gemm_bf16_wmma<<<gp, blk, 0, stream>>>(wk2b, x2t, k2, nullptr, H, C2, N,
      (long long)N * C2, (long long)H * N, 1.f, 1);
  gemm_bf16_wmma<<<gp, blk, 0, stream>>>(wv2b, x2t, v2, nullptr, H, C2, N,
      (long long)N * C2, (long long)H * N, 1.f, 0);

  dim3 ga(N / 128, Bn * 8);  // 18 row-tiles x 16 (batch,head) problems
  flash_attn_bf16<<<ga, blk, 0, stream>>>(q1, k2, v1, o1, N);  // attn12 -> o1
  flash_attn_bf16<<<ga, blk, 0, stream>>>(q2, k1, v2, o2, N);  // attn21 -> o2

  // final projections (f32 out + bias), concatenated into d_out
  dim3 go1(N / 128, (C1 + 127) / 128, Bn);
  gemm_bf16_wmma<<<go1, blk, 0, stream>>>(wo1b, o1, d_out, bo1, C1, H, N,
      (long long)N * H, (long long)C1 * N, 1.f, 2);
  dim3 go2(N / 128, (C2 + 127) / 128, Bn);
  gemm_bf16_wmma<<<go2, blk, 0, stream>>>(wo2b, o2,
      (void*)((float*)d_out + (size_t)Bn * C1 * N), bo2, C2, H, N,
      (long long)N * H, (long long)C2 * N, 1.f, 2);
}
